// GraphAttentionLayer_23837068493279
// MI455X (gfx1250) — matrix operand order repaired and checked by reading, not compiled, not run on hardware
//
#include <hip/hip_runtime.h>
#include <stdint.h>

typedef __attribute__((ext_vector_type(16))) __bf16 v16bf;
typedef __attribute__((ext_vector_type(8)))  float  v8f;

#define N_NODES 8192
#define DIN 512
#define DOUT 512

__device__ __forceinline__ unsigned short f2bf(float f) {
  unsigned int u = __float_as_uint(f);
  u += 0x7FFFu + ((u >> 16) & 1u);   // round-to-nearest-even
  return (unsigned short)(u >> 16);
}
__device__ __forceinline__ float bf2f(unsigned short h) {
  return __uint_as_float(((unsigned int)h) << 16);
}

union Frag {
  v16bf v;
  uint4 q[2];
  unsigned short us[16];
};

// ---------------- conversion kernels ----------------

__global__ void k_cvt_h(const float* __restrict__ h, unsigned short* __restrict__ hbf) {
  size_t i = (size_t)blockIdx.x * blockDim.x + threadIdx.x;
  if (i < (size_t)N_NODES * DIN) hbf[i] = f2bf(h[i]);
}

// Wt[n][k] = bf16(W[k][n]);  W is [DIN][DOUT] row-major
__global__ void k_wt(const float* __restrict__ W, unsigned short* __restrict__ Wt) {
  int k = blockIdx.x * 32 + (threadIdx.x & 31);
  int n = blockIdx.y * 8 + (threadIdx.x >> 5);
  Wt[(size_t)n * DIN + k] = f2bf(W[(size_t)k * DOUT + n]);
}

// ---------------- GEMM1: Wh = h @ W + bW, stored transposed bf16 ----------------
__global__ void __launch_bounds__(256) k_gemm1(
    const unsigned short* __restrict__ hbf,
    const unsigned short* __restrict__ Wt,
    const float* __restrict__ bW,
    unsigned short* __restrict__ WhT) {
  int wave = blockIdx.x * 8 + (threadIdx.x >> 5);
  int lane = threadIdx.x & 31;
  int mt = wave & 511;        // 512 row tiles
  int nt = wave >> 9;         // 32 col tiles
  int M0 = mt * 16, N0 = nt * 16;
  int half = lane >> 4;
  int lrow = lane & 15;
  int n = N0 + lrow;
  const unsigned short* arow = hbf + (size_t)(M0 + lrow) * DIN;
  const unsigned short* brow = Wt + (size_t)n * DIN;
  v8f acc = {};
  for (int k0 = 0; k0 < DIN; k0 += 32) {
    Frag a, b;
    // 16-bit A 16x32 layout: lane half 0 -> K {0..7,16..23}, half 1 -> K {8..15,24..31}
    a.q[0] = *(const uint4*)(arow + k0 + half * 8);
    a.q[1] = *(const uint4*)(arow + k0 + 16 + half * 8);
    // 16-bit B 32x16 layout: lane holds column n, same K order as A: half 0 -> K {0..7,16..23}, half 1 -> K {8..15,24..31}
    b.q[0] = *(const uint4*)(brow + k0 + half * 8);
    b.q[1] = *(const uint4*)(brow + k0 + 16 + half * 8);
    acc = __builtin_amdgcn_wmma_f32_16x16x32_bf16(false, a.v, false, b.v,
                                                  (short)0, acc, false, false);
  }
  float bias = bW[n];
#pragma unroll
  for (int r = 0; r < 8; ++r) {
    int M = M0 + r + 8 * half;           // C layout: lane half selects M+8
    WhT[(size_t)n * N_NODES + M] = f2bf(acc[r] + bias);
  }
}

// ---------------- Wha1/Wha2 matvecs ----------------
__global__ void k_wha(const unsigned short* __restrict__ WhT,
                      const float* __restrict__ a1, const float* __restrict__ b1,
                      const float* __restrict__ a2, const float* __restrict__ b2,
                      float* __restrict__ Wha1, float* __restrict__ Wha2) {
  int i = blockIdx.x * blockDim.x + threadIdx.x;
  float s1 = b1[0], s2 = b2[0];
  for (int nn = 0; nn < DOUT; ++nn) {
    float w = bf2f(WhT[(size_t)nn * N_NODES + i]);
    s1 = fmaf(w, a1[nn], s1);
    s2 = fmaf(w, a2[nn], s2);
  }
  Wha1[i] = s1;
  Wha2[i] = s2;
}

// ---------------- per-row softmax stats: max and 1/sum(exp) ----------------
__global__ void __launch_bounds__(256) k_stats(
    const int* __restrict__ adj,
    const float* __restrict__ Wha1, const float* __restrict__ Wha2,
    float* __restrict__ rmax, float* __restrict__ rsuminv) {
  int row = blockIdx.x * 8 + (threadIdx.x >> 5);
  int lane = threadIdx.x & 31;
  float w1 = Wha1[row];
  const int* arow = adj + (size_t)row * N_NODES;
  float m = -__builtin_inff(), s = 0.f;
  for (int j0 = 0; j0 < N_NODES; j0 += 32) {
    int j = j0 + lane;
    __builtin_prefetch(arow + j + 256, 0, 0);   // stream-ahead: global_prefetch_b8
    float e = w1 + Wha2[j];
    e = e > 0.f ? e : 0.2f * e;            // leaky relu
    e = arow[j] > 0 ? e : -9.0e15f;        // mask
    float mn = fmaxf(m, e);
    s = s * __expf(m - mn) + __expf(e - mn);
    m = mn;
  }
  for (int off = 16; off > 0; off >>= 1) {
    float mo = __shfl_xor(m, off, 32);
    float so = __shfl_xor(s, off, 32);
    float mn = fmaxf(m, mo);
    s = s * __expf(m - mn) + so * __expf(mo - mn);
    m = mn;
  }
  if (lane == 0) { rmax[row] = m; rsuminv[row] = 1.0f / s; }
}

// ---------------- fused attention @ Wh (flash-style, WMMA) ----------------
// Block = 512 threads = 16 waves; block owns 64 rows; wave w owns 32 output cols.
// Each wave: 4 row-tiles x 2 col-tiles of 16x16 accumulators; every B-fragment
// loaded from WhT feeds 4 WMMAs (4x less L2 traffic than 16-row blocking).
__global__ void __launch_bounds__(512) k_attn(
    const int* __restrict__ adj,
    const unsigned short* __restrict__ WhT,
    const float* __restrict__ Wha1, const float* __restrict__ Wha2,
    const float* __restrict__ rmax, const float* __restrict__ rsuminv,
    float* __restrict__ out) {
  __shared__ unsigned short Pt[64 * 32];   // bf16 probability tile (4 KB)
  __shared__ float sw1[64], sm[64], sinv[64];
  int t = threadIdx.x;
  int i0 = blockIdx.x * 64;
  if (t < 64) {
    sw1[t] = Wha1[i0 + t];
    sm[t] = rmax[i0 + t];
    sinv[t] = rsuminv[i0 + t];
  }
  __syncthreads();
  int lane = t & 31;
  int wv = t >> 5;          // 0..15
  int half = lane >> 4;
  int lrow = lane & 15;
  int n0 = wv * 32;
  int pr = t >> 3;          // P-tile row (0..63)
  int pc = (t & 7) * 4;     // P-tile col group (0,4,...,28)
  float w1 = sw1[pr], mm = sm[pr], iv = sinv[pr];
  const int* arow = adj + (size_t)(i0 + pr) * N_NODES;
  const unsigned short* bp0 = WhT + (size_t)(n0 + 0  + lrow) * N_NODES;
  const unsigned short* bp1 = WhT + (size_t)(n0 + 16 + lrow) * N_NODES;
  v8f acc[4][2];
#pragma unroll
  for (int rt = 0; rt < 4; ++rt)
#pragma unroll
    for (int nt = 0; nt < 2; ++nt) acc[rt][nt] = {};

  for (int j0 = 0; j0 < N_NODES; j0 += 32) {
    // cooperative probability tile: each thread computes 4 elements of 64x32
    int4 av = *(const int4*)(arow + j0 + pc);
    __builtin_prefetch(arow + j0 + pc + 128, 0, 0);  // 4 chunks ahead
    float4 w2 = *(const float4*)(Wha2 + j0 + pc);
    float e0 = w1 + w2.x, e1 = w1 + w2.y, e2 = w1 + w2.z, e3 = w1 + w2.w;
    e0 = e0 > 0.f ? e0 : 0.2f * e0;
    e1 = e1 > 0.f ? e1 : 0.2f * e1;
    e2 = e2 > 0.f ? e2 : 0.2f * e2;
    e3 = e3 > 0.f ? e3 : 0.2f * e3;
    e0 = av.x > 0 ? e0 : -9.0e15f;
    e1 = av.y > 0 ? e1 : -9.0e15f;
    e2 = av.z > 0 ? e2 : -9.0e15f;
    e3 = av.w > 0 ? e3 : -9.0e15f;
    float p0 = __expf(e0 - mm) * iv;   // masked entries underflow to exactly 0
    float p1 = __expf(e1 - mm) * iv;
    float p2 = __expf(e2 - mm) * iv;
    float p3 = __expf(e3 - mm) * iv;
    uint2 pk;
    pk.x = (unsigned int)f2bf(p0) | ((unsigned int)f2bf(p1) << 16);
    pk.y = (unsigned int)f2bf(p2) | ((unsigned int)f2bf(p3) << 16);
    *(uint2*)(&Pt[pr * 32 + pc]) = pk;
    __syncthreads();
    // A fragments for the 4 row-tiles from LDS
    Frag a0, a1f, a2f, a3f;
    a0.q[0]  = *(const uint4*)(&Pt[(0 * 16 + lrow) * 32 + half * 8]);
    a0.q[1]  = *(const uint4*)(&Pt[(0 * 16 + lrow) * 32 + 16 + half * 8]);
    a1f.q[0] = *(const uint4*)(&Pt[(1 * 16 + lrow) * 32 + half * 8]);
    a1f.q[1] = *(const uint4*)(&Pt[(1 * 16 + lrow) * 32 + 16 + half * 8]);
    a2f.q[0] = *(const uint4*)(&Pt[(2 * 16 + lrow) * 32 + half * 8]);
    a2f.q[1] = *(const uint4*)(&Pt[(2 * 16 + lrow) * 32 + 16 + half * 8]);
    a3f.q[0] = *(const uint4*)(&Pt[(3 * 16 + lrow) * 32 + half * 8]);
    a3f.q[1] = *(const uint4*)(&Pt[(3 * 16 + lrow) * 32 + 16 + half * 8]);
    __syncthreads();
    // col-tile 0: one B fragment feeds 4 WMMAs
    Frag b;
    b.q[0] = *(const uint4*)(bp0 + j0 + half * 8);
    b.q[1] = *(const uint4*)(bp0 + j0 + 16 + half * 8);
    acc[0][0] = __builtin_amdgcn_wmma_f32_16x16x32_bf16(false, a0.v,  false, b.v, (short)0, acc[0][0], false, false);
    acc[1][0] = __builtin_amdgcn_wmma_f32_16x16x32_bf16(false, a1f.v, false, b.v, (short)0, acc[1][0], false, false);
    acc[2][0] = __builtin_amdgcn_wmma_f32_16x16x32_bf16(false, a2f.v, false, b.v, (short)0, acc[2][0], false, false);
    acc[3][0] = __builtin_amdgcn_wmma_f32_16x16x32_bf16(false, a3f.v, false, b.v, (short)0, acc[3][0], false, false);
    // col-tile 1
    b.q[0] = *(const uint4*)(bp1 + j0 + half * 8);
    b.q[1] = *(const uint4*)(bp1 + j0 + 16 + half * 8);
    acc[0][1] = __builtin_amdgcn_wmma_f32_16x16x32_bf16(false, a0.v,  false, b.v, (short)0, acc[0][1], false, false);
    acc[1][1] = __builtin_amdgcn_wmma_f32_16x16x32_bf16(false, a1f.v, false, b.v, (short)0, acc[1][1], false, false);
    acc[2][1] = __builtin_amdgcn_wmma_f32_16x16x32_bf16(false, a2f.v, false, b.v, (short)0, acc[2][1], false, false);
    acc[3][1] = __builtin_amdgcn_wmma_f32_16x16x32_bf16(false, a3f.v, false, b.v, (short)0, acc[3][1], false, false);
  }
#pragma unroll
  for (int rt = 0; rt < 4; ++rt)
#pragma unroll
    for (int nt = 0; nt < 2; ++nt)
#pragma unroll
      for (int r = 0; r < 8; ++r) {
        size_t orow = (size_t)(i0 + rt * 16 + r + 8 * half) * DOUT;
        out[orow + n0 + nt * 16 + lrow] = acc[rt][nt][r];
      }
}

// ---------------- launch ----------------
extern "C" void kernel_launch(void* const* d_in, const int* in_sizes, int n_in,
                              void* d_out, int out_size, void* d_ws, size_t ws_size,
                              hipStream_t stream) {
  const float* h   = (const float*)d_in[0];
  const int*   adj = (const int*)d_in[1];
  const float* W   = (const float*)d_in[2];
  const float* bW  = (const float*)d_in[3];
  const float* a1  = (const float*)d_in[4];
  const float* b1  = (const float*)d_in[5];
  const float* a2  = (const float*)d_in[6];
  const float* b2  = (const float*)d_in[7];
  float* out = (float*)d_out;

  char* ws = (char*)d_ws;
  // workspace layout (all offsets 256B-aligned), total ~16.6 MB
  unsigned short* hbf  = (unsigned short*)(ws + 0);          // 8 MB
  unsigned short* Wt   = (unsigned short*)(ws + 8388608);    // 512 KB
  unsigned short* WhT  = (unsigned short*)(ws + 8912896);    // 8 MB
  float* Wha1 = (float*)(ws + 17301504);                     // 32 KB
  float* Wha2 = (float*)(ws + 17334272);                     // 32 KB
  float* rmax = (float*)(ws + 17367040);                     // 32 KB
  float* rsin = (float*)(ws + 17399808);                     // 32 KB
  (void)in_sizes; (void)n_in; (void)out_size; (void)ws_size;

  k_cvt_h<<<(N_NODES * DIN) / 256, 256, 0, stream>>>(h, hbf);
  k_wt<<<dim3(DIN / 32, DOUT / 8), 256, 0, stream>>>(W, Wt);
  k_gemm1<<<(N_NODES / 16) * (DOUT / 16) / 8, 256, 0, stream>>>(hbf, Wt, bW, WhT);
  k_wha<<<N_NODES / 256, 256, 0, stream>>>(WhT, a1, b1, a2, b2, Wha1, Wha2);
  k_stats<<<N_NODES / 8, 256, 0, stream>>>(adj, Wha1, Wha2, rmax, rsin);
  k_attn<<<N_NODES / 64, 512, 0, stream>>>(adj, WhT, Wha1, Wha2, rmax, rsin, out);
}
